// GatedDeltaRule_77000173682703
// MI455X (gfx1250) — matrix operand order repaired and checked by reading, not compiled, not run on hardware
//
#include <hip/hip_runtime.h>
#include <hip/hip_bf16.h>
#include <cstdint>

#define HID  1024
#define NH   16
#define HD   64
#define BB   4
#define TT   2048
#define NROW (BB*TT)   // 8192

typedef __attribute__((ext_vector_type(16))) __bf16 v16bf;
typedef __attribute__((ext_vector_type(8)))  __bf16 v8bf;
typedef __attribute__((ext_vector_type(8)))  float  v8f;

// float -> bf16, round-to-nearest-even, via integer ops
static __device__ __forceinline__ __bf16 f2bf(float f) {
    union { float f; uint32_t u; } cv; cv.f = f;
    uint32_t r = cv.u + 0x7FFFu + ((cv.u >> 16) & 1u);
    union { uint16_t s; __bf16 b; } cb; cb.s = (uint16_t)(r >> 16);
    return cb.b;
}

static __device__ __forceinline__ void wait_async0() {
#if __has_builtin(__builtin_amdgcn_s_wait_asynccnt)
    __builtin_amdgcn_s_wait_asynccnt(0);
#else
    asm volatile("s_wait_asynccnt 0x0" ::: "memory");
#endif
}

// CDNA5 async copy: 16 bytes global -> LDS, tracked by ASYNCcnt.
static __device__ __forceinline__ void async_cp16(unsigned lds_addr, const void* gptr) {
    asm volatile("global_load_async_to_lds_b128 %0, %1, off"
                 :: "v"(lds_addr), "v"(gptr) : "memory");
}

// Scheduler steering: issue groups in order. Builtin needs constant-int args,
// so pass group sizes as template parameters.
template <int NVM, int NDS, int NWMMA>
static __device__ __forceinline__ void sched_vmem_ds_wmma() {
#if __has_builtin(__builtin_amdgcn_sched_group_barrier)
    if constexpr (NVM > 0)
        __builtin_amdgcn_sched_group_barrier(0x020, NVM, 0);   // VMEM reads
    if constexpr (NDS > 0)
        __builtin_amdgcn_sched_group_barrier(0x100, NDS, 0);   // DS reads
    if constexpr (NWMMA > 0)
        __builtin_amdgcn_sched_group_barrier(0x008, NWMMA, 0); // WMMA
#endif
}

__global__ void cvt_f32_bf16(const float* __restrict__ in, __bf16* __restrict__ out, int n) {
    int i = blockIdx.x * 256 + threadIdx.x;
    if (i < n) out[i] = f2bf(in[i]);
}

// C[N, NC] = A[N,K] * W[NC,K]^T   (A,W bf16 row-major, C f32 row-major)
// Block = 256 threads = 8 waves; block tile 128 rows x 64 cols.
// W tile (64x32, 4KB) shared by all 8 waves: LDS double buffer filled by
// global_load_async_to_lds_b128 one K-step ahead (ASYNCcnt). A fragment is
// register double-buffered. Last K-step peeled -> branch-free steady state.
#define WPAD 40   // padded LDS row stride (elements) to spread banks
__global__ __launch_bounds__(256)
void gemm_bf16_wmma(const __bf16* __restrict__ A, const __bf16* __restrict__ W,
                    float* __restrict__ C, int N, int K, int NC) {
    (void)N;
    __shared__ __align__(16) __bf16 tw[2][64][WPAD];

    const int tid  = threadIdx.x;
    const int lane = tid & 31;
    const int wave = tid >> 5;                 // 0..7
    const int half = lane >> 4;                // 0/1
    const int l16  = lane & 15;
    const int m0 = blockIdx.y * 128 + wave * 16;
    const int n0 = blockIdx.x * 64;
    const int m  = m0 + l16;

    // cooperative copy assignment: thread -> (row in tile, 8-elem K segment)
    const int cn = tid >> 2;                   // 0..63
    const int cs = (tid & 3) * 8;              // 0,8,16,24
    const __bf16* wsrc = W + (size_t)(n0 + cn) * K + cs;
    const unsigned ldst0 = (unsigned)(size_t)&tw[0][cn][cs];
    const unsigned ldst1 = (unsigned)(size_t)&tw[1][cn][cs];

    v8f acc[4] = {};
    const __bf16* arow = A + (size_t)m * K;

    // prologue: async-fill buffer 0 (K-step 0); A fragment 0 into registers
    async_cp16(ldst0, wsrc);
    v8bf alo = *(const v8bf*)(arow + half * 8);
    v8bf ahi = *(const v8bf*)(arow + half * 8 + 16);
    wait_async0();
    __syncthreads();

    int kk = 0;
    for (; kk < K - 32; kk += 32) {
        const int cur = (kk >> 5) & 1;
        // async-fill the other buffer for K-step kk+32
        async_cp16(cur ? ldst0 : ldst1, wsrc + kk + 32);

        // next A fragment into registers (consumed next iteration)
        const __bf16* apn = arow + kk + 32 + half * 8;
        __builtin_prefetch(apn + 32, 0, 1);          // speculative, 2 steps ahead
        v8bf alo_n = *(const v8bf*)(apn);
        v8bf ahi_n = *(const v8bf*)(apn + 16);

        // all 4 B fragments from LDS into distinct registers
        v16bf bfr[4];
#pragma unroll
        for (int tn = 0; tn < 4; ++tn) {
            const __bf16* bp = &tw[cur][tn * 16 + l16][half * 8];
            v8bf blo = *(const v8bf*)(bp);
            v8bf bhi = *(const v8bf*)(bp + 16);
#pragma unroll
            for (int i = 0; i < 8; ++i) { bfr[tn][i] = blo[i]; bfr[tn][i + 8] = bhi[i]; }
        }

        v16bf a;
#pragma unroll
        for (int i = 0; i < 8; ++i) { a[i] = alo[i]; a[i + 8] = ahi[i]; }

#pragma unroll
        for (int tn = 0; tn < 4; ++tn)
            acc[tn] = __builtin_amdgcn_wmma_f32_16x16x32_bf16(
                false, a, false, bfr[tn], (short)0, acc[tn], false, false);

        // force: 2 A-loads -> 8 B ds_loads -> 4 back-to-back WMMAs
        sched_vmem_ds_wmma<2, 8, 4>();

        alo = alo_n; ahi = ahi_n;
        wait_async0();      // next tile resident
        __syncthreads();    // all waves done reading cur before it is refilled
    }

    // epilogue: last K-step, everything already resident
    {
        const int cur = (kk >> 5) & 1;
        v16bf bfr[4];
#pragma unroll
        for (int tn = 0; tn < 4; ++tn) {
            const __bf16* bp = &tw[cur][tn * 16 + l16][half * 8];
            v8bf blo = *(const v8bf*)(bp);
            v8bf bhi = *(const v8bf*)(bp + 16);
#pragma unroll
            for (int i = 0; i < 8; ++i) { bfr[tn][i] = blo[i]; bfr[tn][i + 8] = bhi[i]; }
        }
        v16bf a;
#pragma unroll
        for (int i = 0; i < 8; ++i) { a[i] = alo[i]; a[i + 8] = ahi[i]; }
#pragma unroll
        for (int tn = 0; tn < 4; ++tn)
            acc[tn] = __builtin_amdgcn_wmma_f32_16x16x32_bf16(
                false, a, false, bfr[tn], (short)0, acc[tn], false, false);
        sched_vmem_ds_wmma<0, 8, 4>();
    }

#pragma unroll
    for (int tn = 0; tn < 4; ++tn) {
        const int n = n0 + tn * 16 + l16;
#pragma unroll
        for (int r = 0; r < 8; ++r)
            C[(size_t)(m0 + r + 8 * half) * NC + n] = acc[tn][r];
    }
}

// alpha = sigmoid(x @ Wa^T + ba), beta = sigmoid(x @ Wb^T + bb)
__global__ void ab_proj(const float* __restrict__ x,
                        const float* __restrict__ Wa, const float* __restrict__ ba,
                        const float* __restrict__ Wb, const float* __restrict__ bb,
                        float* __restrict__ alpha, float* __restrict__ beta) {
    int idx = blockIdx.x * 256 + threadIdx.x;
    if (idx >= NROW * NH * 2) return;
    int r = idx >> 5;
    int rem = idx & 31;
    int h = rem & 15;
    int which = rem >> 4;
    const float4* xr = (const float4*)(x + (size_t)r * HID);
    const float4* wr = (const float4*)((which ? Wb : Wa) + (size_t)h * HID);
    float s = 0.f;
    for (int kq = 0; kq < HID / 4; ++kq) {
        float4 a = xr[kq], w = wr[kq];
        s += a.x * w.x + a.y * w.y + a.z * w.z + a.w * w.w;
    }
    s += (which ? bb[h] : ba[h]);
    float sg = 1.f / (1.f + __expf(-s));
    (which ? beta : alpha)[(size_t)r * NH + h] = sg;
}

// per-head L2 norm over D=64, in place
__global__ void l2norm_rows64(float* __restrict__ p, int nrows) {
    int r = blockIdx.x * 256 + threadIdx.x;
    if (r >= nrows) return;
    float4* row = (float4*)(p + (size_t)r * HD);
    float s = 0.f;
#pragma unroll
    for (int i = 0; i < 16; ++i) {
        float4 v = row[i];
        s += v.x * v.x + v.y * v.y + v.z * v.z + v.w * v.w;
    }
    float inv = 1.f / fmaxf(sqrtf(s), 1e-6f);
#pragma unroll
    for (int i = 0; i < 16; ++i) {
        float4 v = row[i];
        v.x *= inv; v.y *= inv; v.z *= inv; v.w *= inv;
        row[i] = v;
    }
}

__global__ void silu_inplace(float* __restrict__ g, int n) {
    int i = blockIdx.x * 256 + threadIdx.x;
    if (i < n) { float v = g[i]; g[i] = v / (1.f + __expf(-v)); }
}

// Gated delta-rule scan. One block per (b,h); thread i owns state row S[i][0..63].
__global__ __launch_bounds__(64)
void scan_kernel(const float* __restrict__ q, const float* __restrict__ k,
                 const float* __restrict__ v, const float* __restrict__ alpha,
                 const float* __restrict__ beta, const float* __restrict__ gate,
                 __bf16* __restrict__ og, float* __restrict__ Sout) {
    __shared__ __align__(16) float ql[HD], kl[HD], vl[HD];
    const int b = blockIdx.x / NH;
    const int h = blockIdx.x % NH;
    const int i = threadIdx.x;
    float S[HD];
#pragma unroll
    for (int j = 0; j < HD; ++j) S[j] = 0.f;

    for (int t = 0; t < TT; ++t) {
        const size_t row = (size_t)(b * TT + t);
        const size_t vidx = row * HID + h * HD + i;
        ql[i] = q[vidx]; kl[i] = k[vidx]; vl[i] = v[vidx];
        __syncthreads();
        const float at = alpha[row * NH + h];
        const float bt = beta[row * NH + h];
        const float4* k4 = (const float4*)kl;
        const float4* q4 = (const float4*)ql;
        float Sk = 0.f;
#pragma unroll
        for (int jq = 0; jq < 16; ++jq) {
            float4 kv = k4[jq];
            Sk += S[4*jq+0]*kv.x + S[4*jq+1]*kv.y + S[4*jq+2]*kv.z + S[4*jq+3]*kv.w;
        }
        const float ci = bt * vl[i] - at * bt * Sk;
        float o = 0.f;
#pragma unroll
        for (int jq = 0; jq < 16; ++jq) {
            float4 kv = k4[jq], qv = q4[jq];
            S[4*jq+0] = at*S[4*jq+0] + ci*kv.x; o += S[4*jq+0]*qv.x;
            S[4*jq+1] = at*S[4*jq+1] + ci*kv.y; o += S[4*jq+1]*qv.y;
            S[4*jq+2] = at*S[4*jq+2] + ci*kv.z; o += S[4*jq+2]*qv.z;
            S[4*jq+3] = at*S[4*jq+3] + ci*kv.w; o += S[4*jq+3]*qv.w;
        }
        og[vidx] = f2bf(o * gate[vidx]);
        __syncthreads();
    }
    float4* dst = (float4*)(Sout + (((size_t)b * NH + h) * HD + i) * HD);
#pragma unroll
    for (int jq = 0; jq < 16; ++jq)
        dst[jq] = make_float4(S[4*jq], S[4*jq+1], S[4*jq+2], S[4*jq+3]);
}

extern "C" void kernel_launch(void* const* d_in, const int* in_sizes, int n_in,
                              void* d_out, int out_size, void* d_ws, size_t ws_size,
                              hipStream_t stream) {
    (void)in_sizes; (void)n_in; (void)out_size; (void)ws_size;
    const float* x  = (const float*)d_in[0];
    const float* Wq = (const float*)d_in[1];
    const float* Wk = (const float*)d_in[2];
    const float* Wv = (const float*)d_in[3];
    const float* Wa = (const float*)d_in[4];
    const float* ba = (const float*)d_in[5];
    const float* Wb = (const float*)d_in[6];
    const float* bb = (const float*)d_in[7];
    const float* Wg = (const float*)d_in[8];
    const float* Wo = (const float*)d_in[9];
    float* out = (float*)d_out;                 // [NROW*HID] o, then [B*H*D*D] S_new

    char* p = (char*)d_ws;
    auto take = [&](size_t bytes) { void* q = (void*)p; p += (bytes + 255) & ~(size_t)255; return q; };
    __bf16* xb  = (__bf16*)take((size_t)NROW * HID * 2);
    __bf16* wqb = (__bf16*)take((size_t)HID * HID * 2);
    __bf16* wkb = (__bf16*)take((size_t)HID * HID * 2);
    __bf16* wvb = (__bf16*)take((size_t)HID * HID * 2);
    __bf16* wgb = (__bf16*)take((size_t)HID * HID * 2);
    __bf16* wob = (__bf16*)take((size_t)HID * HID * 2);
    float*  qh  = (float*)take((size_t)NROW * HID * 4);
    float*  kh  = (float*)take((size_t)NROW * HID * 4);
    float*  vh  = (float*)take((size_t)NROW * HID * 4);
    float*  gh  = (float*)take((size_t)NROW * HID * 4);
    float*  al  = (float*)take((size_t)NROW * NH * 4);
    float*  be  = (float*)take((size_t)NROW * NH * 4);
    __bf16* og  = (__bf16*)take((size_t)NROW * HID * 2);

    const int ne = NROW * HID;
    const int nw = HID * HID;
    cvt_f32_bf16<<<(ne + 255) / 256, 256, 0, stream>>>(x,  xb,  ne);
    cvt_f32_bf16<<<(nw + 255) / 256, 256, 0, stream>>>(Wq, wqb, nw);
    cvt_f32_bf16<<<(nw + 255) / 256, 256, 0, stream>>>(Wk, wkb, nw);
    cvt_f32_bf16<<<(nw + 255) / 256, 256, 0, stream>>>(Wv, wvb, nw);
    cvt_f32_bf16<<<(nw + 255) / 256, 256, 0, stream>>>(Wg, wgb, nw);
    cvt_f32_bf16<<<(nw + 255) / 256, 256, 0, stream>>>(Wo, wob, nw);

    dim3 gg(HID / 64, NROW / 128);   // (16, 64)
    gemm_bf16_wmma<<<gg, 256, 0, stream>>>(xb, wqb, qh, NROW, HID, HID);
    gemm_bf16_wmma<<<gg, 256, 0, stream>>>(xb, wkb, kh, NROW, HID, HID);
    gemm_bf16_wmma<<<gg, 256, 0, stream>>>(xb, wvb, vh, NROW, HID, HID);
    gemm_bf16_wmma<<<gg, 256, 0, stream>>>(xb, wgb, gh, NROW, HID, HID);

    ab_proj<<<(NROW * NH * 2 + 255) / 256, 256, 0, stream>>>(x, Wa, ba, Wb, bb, al, be);

    l2norm_rows64<<<(NROW * NH + 255) / 256, 256, 0, stream>>>(qh, NROW * NH);
    l2norm_rows64<<<(NROW * NH + 255) / 256, 256, 0, stream>>>(kh, NROW * NH);
    silu_inplace<<<(ne + 255) / 256, 256, 0, stream>>>(gh, ne);

    scan_kernel<<<BB * NH, HD, 0, stream>>>(qh, kh, vh, al, be, gh, og,
                                            out + (size_t)NROW * HID);

    gemm_bf16_wmma<<<gg, 256, 0, stream>>>(og, wob, out, NROW, HID, HID);
}
